// SpMiddleResNetFHD_45294725103947
// MI455X (gfx1250) — compile-verified, matrix operand order checked by reading, not run on hardware
//
#include <hip/hip_runtime.h>
#include <hip/hip_bf16.h>
#include <stdint.h>

// ---------------------------------------------------------------------------
// Types
// ---------------------------------------------------------------------------
typedef __attribute__((ext_vector_type(16))) __bf16        v16bf;
typedef __attribute__((ext_vector_type(8)))  float         v8f;
typedef __attribute__((ext_vector_type(4)))  unsigned int  v4u;

typedef unsigned short bf16_t;

union AFrag { v16bf v; v4u q[2]; };

__device__ __forceinline__ float bf2f(bf16_t h) {
  union { uint32_t i; float f; } x; x.i = ((uint32_t)h) << 16; return x.f;
}
__device__ __forceinline__ bf16_t f2bf(float f) {
  union { float f; uint32_t i; } x; x.f = f;
  uint32_t r = x.i + 0x7FFFu + ((x.i >> 16) & 1u);   // round-to-nearest-even
  return (bf16_t)(r >> 16);
}

// ---------------------------------------------------------------------------
// Fused implicit-GEMM conv3d + bias + BN + mask + residual + ReLU (bf16 WMMA)
//
// One wave32 computes a 32-voxel x 16-out-channel tile (two accumulators
// sharing each B fragment).  Tap geometry is compile-time (template).
//
// Interior fast path: if the block's whole input halo is in-bounds (a
// wave-uniform check, true for the vast majority of blocks), run a loop with
// unconditional loads -- no per-lane guards, no EXEC juggling: the interior
// loop is just 2x b128 (B) + 4x b128 (A) -> 2x v_wmma_f32_16x16x32_bf16.
// Boundary blocks take the guarded loop (per-lane cndmask'd loads + uniform
// tap skipping).
//
// MODE16 packs two taps into one K=32 WMMA (K 0..15 = even tap's 16 channels,
// K 16..31 = odd tap's), so the FLOP-dominant 16-channel full-resolution
// stage runs at full WMMA utilization.
//
// Grid = (wt*cochunks, OH, OD) with wt = OW/32 a power of two at every level,
// so the block-index decode is shifts/masks only (no integer division).
// ---------------------------------------------------------------------------
struct ConvP {
  const bf16_t* in;      // NDHWC bf16, CS channels per voxel
  const bf16_t* wf;      // pre-swizzled B fragments (512 bf16 per fragment)
  const float*  bias;    // [Cout] or null
  const float*  scale;   // BN scale (gamma*rsqrt(var+eps))
  const float*  shift;   // BN shift (beta - mean*scale)
  const bf16_t* mask;    // out-resolution occupancy mask (bf16 0/1) or null
  const bf16_t* skip;    // residual tensor (same shape as out) or null
  bf16_t*       out;
  int D, H, W, OH, OW;
  int Cout, cochunks, lgwt;
  int sd, sh, sw, pd, ph, pw;
};

#define WMMA_BF16(A, B, C) \
  __builtin_amdgcn_wmma_f32_16x16x32_bf16(false, (A), false, (B), (short)0, (C), false, false)

template <int MODE16, int KD, int KH, int KW, int KC, bool GUARD>
__device__ __forceinline__ void conv_loop(
    const ConvP& p, const bf16_t* __restrict__ wfb, size_t wstep,
    int idb, int ihb, int iw0b, int iw1b, int chb, v8f& acc0, v8f& acc1) {
  constexpr int TAPS = KD * KH * KW;
  constexpr int CS   = MODE16 ? 16 : KC * 32;
  const v4u z4 = {0u, 0u, 0u, 0u};

  if constexpr (MODE16) {
    constexpr int NSTEP = (TAPS + 1) / 2;
#pragma unroll
    for (int s = 0; s < NSTEP; ++s) {
      constexpr_hint:;
      const int t0 = 2 * s, t1 = 2 * s + 1;
      const int kd0 = t0 / (KH * KW), kh0 = (t0 / KW) % KH, kw0 = t0 % KW;
      const int kd1 = t1 / (KH * KW), kh1 = (t1 / KW) % KH, kw1 = t1 % KW;
      const bool t1in = (t1 < TAPS);                 // compile-time after unroll
      const int id0 = idb + kd0, ih0 = ihb + kh0;
      const int id1 = idb + kd1, ih1 = ihb + kh1;
      bool u0 = true, u1 = t1in;
      if constexpr (GUARD) {
        u0 = ((uint32_t)id0 < (uint32_t)p.D) & ((uint32_t)ih0 < (uint32_t)p.H);
        u1 = t1in && (((uint32_t)id1 < (uint32_t)p.D) & ((uint32_t)ih1 < (uint32_t)p.H));
        if (!(u0 | u1)) continue;                    // wave-uniform: skip whole step
      }
      const int row0 = (id0 * p.H + ih0) * p.W;
      const int row1 = (id1 * p.H + ih1) * p.W;

      const v4u* wp = (const v4u*)(wfb + (size_t)s * wstep);
      AFrag b; b.q[0] = wp[0]; b.q[1] = wp[1];
      __builtin_prefetch((const void*)(wfb + (size_t)(s + 1) * wstep), 0, 3);

      const int iwA0 = iw0b + kw0, iwA1 = iw0b + kw1;
      const int iwB0 = iw1b + kw0, iwB1 = iw1b + kw1;
      AFrag a0, a1;
      if constexpr (GUARD) {
        a0.q[0] = (u0 & ((uint32_t)iwA0 < (uint32_t)p.W))
                ? *(const v4u*)(p.in + (size_t)(row0 + iwA0) * CS + chb) : z4;
        a0.q[1] = (u1 & ((uint32_t)iwA1 < (uint32_t)p.W))
                ? *(const v4u*)(p.in + (size_t)(row1 + iwA1) * CS + chb) : z4;
        a1.q[0] = (u0 & ((uint32_t)iwB0 < (uint32_t)p.W))
                ? *(const v4u*)(p.in + (size_t)(row0 + iwB0) * CS + chb) : z4;
        a1.q[1] = (u1 & ((uint32_t)iwB1 < (uint32_t)p.W))
                ? *(const v4u*)(p.in + (size_t)(row1 + iwB1) * CS + chb) : z4;
      } else {
        a0.q[0] = *(const v4u*)(p.in + (size_t)(row0 + iwA0) * CS + chb);
        a0.q[1] = t1in ? *(const v4u*)(p.in + (size_t)(row1 + iwA1) * CS + chb) : z4;
        a1.q[0] = *(const v4u*)(p.in + (size_t)(row0 + iwB0) * CS + chb);
        a1.q[1] = t1in ? *(const v4u*)(p.in + (size_t)(row1 + iwB1) * CS + chb) : z4;
      }
      acc0 = WMMA_BF16(a0.v, b.v, acc0);
      acc1 = WMMA_BF16(a1.v, b.v, acc1);
    }
  } else {
#pragma unroll
    for (int t = 0; t < TAPS; ++t) {
      const int kdi = t / (KH * KW), khi = (t / KW) % KH, kwi = t % KW;
      const int id = idb + kdi, ih = ihb + khi;
      if constexpr (GUARD) {
        if (!(((uint32_t)id < (uint32_t)p.D) & ((uint32_t)ih < (uint32_t)p.H)))
          continue;                                  // wave-uniform: skip whole tap
      }
      const int row = (id * p.H + ih) * p.W;
      const int iwA = iw0b + kwi, iwB = iw1b + kwi;
      bool okA = true, okB = true;
      if constexpr (GUARD) {
        okA = (uint32_t)iwA < (uint32_t)p.W;
        okB = (uint32_t)iwB < (uint32_t)p.W;
      }
      const bf16_t* rA = p.in + (size_t)(row + iwA) * CS;
      const bf16_t* rB = p.in + (size_t)(row + iwB) * CS;
      __builtin_prefetch((const void*)(wfb + (size_t)((t + 1) * KC) * wstep), 0, 3);
#pragma unroll
      for (int kc = 0; kc < KC; ++kc) {
        const int s = t * KC + kc;
        const v4u* wp = (const v4u*)(wfb + (size_t)s * wstep);
        AFrag b; b.q[0] = wp[0]; b.q[1] = wp[1];
        AFrag a0, a1;
        if constexpr (GUARD) {
          a0.q[0] = okA ? *(const v4u*)(rA + kc * 32 + chb)      : z4;
          a0.q[1] = okA ? *(const v4u*)(rA + kc * 32 + chb + 16) : z4;
          a1.q[0] = okB ? *(const v4u*)(rB + kc * 32 + chb)      : z4;
          a1.q[1] = okB ? *(const v4u*)(rB + kc * 32 + chb + 16) : z4;
        } else {
          a0.q[0] = *(const v4u*)(rA + kc * 32 + chb);
          a0.q[1] = *(const v4u*)(rA + kc * 32 + chb + 16);
          a1.q[0] = *(const v4u*)(rB + kc * 32 + chb);
          a1.q[1] = *(const v4u*)(rB + kc * 32 + chb + 16);
        }
        acc0 = WMMA_BF16(a0.v, b.v, acc0);
        acc1 = WMMA_BF16(a1.v, b.v, acc1);
      }
    }
  }
}

template <int MODE16, int KD, int KH, int KW, int KC>
__global__ __launch_bounds__(32) void conv_wmma_t(ConvP p) {
  const int lane = (int)threadIdx.x;
  // Block decode: shifts/masks only (wt = OW/32 is a power of two)
  const int bx   = (int)blockIdx.x;
  const int ow0  = (bx & ((1 << p.lgwt) - 1)) << 5;
  const int coch = bx >> p.lgwt;
  const int oh   = (int)blockIdx.y;
  const int od   = (int)blockIdx.z;
  const int co0  = coch << 4;

  // A-fragment lane mapping (16-bit A 16x32): lane<16 -> row=lane, K 0..7/16..23
  //                                           lane>=16 -> row=lane-16, K 8..15/24..31
  const int mrow = lane & 15;
  const int chb  = (lane < 16) ? 0 : 8;
  const int iw0b = (ow0 + mrow) * p.sw - p.pw;    // per-lane input-w base, subtile 0
  const int iw1b = iw0b + 16 * p.sw;              // subtile 1
  const int idb  = od * p.sd - p.pd;
  const int ihb  = oh * p.sh - p.ph;

  const bf16_t* wfb  = p.wf + (((size_t)coch) << 9) + ((size_t)lane << 4);
  const size_t wstep = (size_t)p.cochunks << 9;   // bf16 elements per step

  v8f acc0 = {0.f, 0.f, 0.f, 0.f, 0.f, 0.f, 0.f, 0.f};
  v8f acc1 = acc0;

  // Wave-uniform interior test: whole KD x KH x KW halo of the 32-voxel tile
  // in bounds -> guard-free loop.
  const bool interior =
      (idb >= 0) & (idb + KD <= p.D) &
      (ihb >= 0) & (ihb + KH <= p.H) &
      (ow0 * p.sw - p.pw >= 0) &
      ((ow0 + 31) * p.sw - p.pw + KW <= p.W);

  if (interior)
    conv_loop<MODE16, KD, KH, KW, KC, false>(p, wfb, wstep, idb, ihb, iw0b, iw1b, chb, acc0, acc1);
  else
    conv_loop<MODE16, KD, KH, KW, KC, true >(p, wfb, wstep, idb, ihb, iw0b, iw1b, chb, acc0, acc1);

  // --- Epilogue: C/D layout  m = r + 8*(lane>=16), n = lane&15 ------------
  const int n = lane & 15;
  const int c = co0 + n;
  const float sc = p.scale ? p.scale[c] : 1.0f;
  const float sh = p.shift ? p.shift[c] : 0.0f;
  const float bi = p.bias  ? p.bias[c]  : 0.0f;
  const int mofs = (lane >> 4) << 3;
  const size_t rowv = ((size_t)(od * p.OH + oh)) * (size_t)p.OW;
#pragma unroll
  for (int r = 0; r < 8; ++r) {
    const int m = r + mofs;
    {
      const size_t vox = rowv + (size_t)(ow0 + m);
      float v = acc0[r] + bi;
      v = v * sc + sh;
      if (p.mask) v *= bf2f(p.mask[vox]);
      if (p.skip) v += bf2f(p.skip[vox * (size_t)p.Cout + c]);
      p.out[vox * (size_t)p.Cout + c] = f2bf(fmaxf(v, 0.0f));
    }
    {
      const size_t vox = rowv + (size_t)(ow0 + 16 + m);
      float v = acc1[r] + bi;
      v = v * sc + sh;
      if (p.mask) v *= bf2f(p.mask[vox]);
      if (p.skip) v += bf2f(p.skip[vox * (size_t)p.Cout + c]);
      p.out[vox * (size_t)p.Cout + c] = f2bf(fmaxf(v, 0.0f));
    }
  }
}

// ---------------------------------------------------------------------------
// Weight pre-swizzle: fp32 (kd,kh,kw,ci,co) -> bf16 B-fragments laid out
// exactly as the WMMA lanes consume them (lane-major, 16 elements/lane).
// ---------------------------------------------------------------------------
__global__ void wprep_k(const float* src, bf16_t* dst, int taps, int Ci, int Co,
                        int mode16, int kchunks, int cochunks, int nsteps) {
  const int idx = blockIdx.x * blockDim.x + threadIdx.x;
  const int total = nsteps * cochunks * 512;
  if (idx >= total) return;
  const int j    = idx & 15;
  const int lane = (idx >> 4) & 31;
  const int frag = idx >> 9;
  const int coch = frag % cochunks;
  const int s    = frag / cochunks;
  const int klocal = (j & 7) + ((j & 8) ? 16 : 0) + ((lane >= 16) ? 8 : 0);
  int tap, ci;
  if (mode16) { tap = 2 * s + ((klocal >= 16) ? 1 : 0); ci = klocal & 15; }
  else        { tap = s / kchunks;                      ci = (s % kchunks) * 32 + klocal; }
  const int co = coch * 16 + (lane & 15);
  float v = 0.0f;
  if (tap < taps && ci < Ci && co < Co)
    v = src[((size_t)tap * Ci + ci) * Co + co];
  dst[idx] = f2bf(v);
}

// BN folding: scale = gamma*rsqrt(var+eps), shift = beta - mean*scale
__global__ void bnprep_k(const float* g, const float* b, const float* m, const float* v,
                         float* scale, float* shift, int C) {
  const int i = blockIdx.x * blockDim.x + threadIdx.x;
  if (i >= C) return;
  const float s = g[i] * rsqrtf(v[i] + 1e-3f);
  scale[i] = s;
  shift[i] = b[i] - m[i] * s;
}

// Scatter voxel features into dense bf16 grid (Cin=5 padded to 16) + mask
__global__ void scatter_k(const float* vf, const int* coors, bf16_t* grid,
                          bf16_t* mask, int Nv, int H, int W) {
  const int i = blockIdx.x * blockDim.x + threadIdx.x;
  if (i >= Nv) return;
  const int z = coors[i * 4 + 1], y = coors[i * 4 + 2], x = coors[i * 4 + 3];
  const size_t vox = ((size_t)z * H + y) * W + x;
#pragma unroll
  for (int c = 0; c < 5; ++c) grid[vox * 16 + c] = f2bf(vf[i * 5 + c]);
  mask[vox] = (bf16_t)0x3F80;  // 1.0bf16
}

// mask_down = (conv(mask, ones) > 0)
__global__ void maskpool_k(const bf16_t* mi, bf16_t* mo,
                           int D, int H, int W, int OD, int OH, int OW,
                           int kd, int kh, int kw, int sd, int sh, int sw,
                           int pd, int ph, int pw) {
  const int idx = blockIdx.x * blockDim.x + threadIdx.x;
  const int total = OD * OH * OW;
  if (idx >= total) return;
  const int ow = idx % OW; int t = idx / OW;
  const int oh = t % OH;   const int od = t / OH;
  int any = 0;
  for (int a = 0; a < kd; ++a) {
    const int id = od * sd - pd + a; if ((uint32_t)id >= (uint32_t)D) continue;
    for (int b = 0; b < kh; ++b) {
      const int ih = oh * sh - ph + b; if ((uint32_t)ih >= (uint32_t)H) continue;
      for (int c = 0; c < kw; ++c) {
        const int iw = ow * sw - pw + c; if ((uint32_t)iw >= (uint32_t)W) continue;
        any |= (mi[((size_t)id * H + ih) * W + iw] != 0);
      }
    }
  }
  mo[idx] = any ? (bf16_t)0x3F80 : (bf16_t)0;
}

// (1,H,W,C) bf16 -> (C,H,W) fp32
__global__ void finalize_k(const bf16_t* x, float* out, int C, int HW) {
  const int idx = blockIdx.x * blockDim.x + threadIdx.x;
  if (idx >= C * HW) return;
  const int c = idx / HW, rem = idx % HW;
  out[idx] = bf2f(x[(size_t)rem * C + c]);
}

// ---------------------------------------------------------------------------
// Host orchestration
// ---------------------------------------------------------------------------
static inline int csz_host(int Ci) { return (Ci <= 16) ? 16 : ((Ci + 31) / 32) * 32; }
static inline int lg2_host(int x)  { int l = 0; while ((1 << l) < x) ++l; return l; }

extern "C" void kernel_launch(void* const* d_in, const int* in_sizes, int n_in,
                              void* d_out, int out_size, void* d_ws, size_t ws_size,
                              hipStream_t stream) {
  (void)n_in; (void)out_size; (void)ws_size;
  const float* VF    = (const float*)d_in[0];
  const int*   COORS = (const int*)d_in[1];
  const int    NV    = in_sizes[0] / 5;

  // Level dims (fixed by the reference's input_shape = [256,256,24])
  const int D1 = 25, H1 = 256, W1 = 256;
  const int D2 = 13, H2 = 128, W2 = 128;
  const int D3 = 7,  H3 = 64,  W3 = 64;
  const int D4 = 3,  H4 = 32,  W4 = 32;
  const int D5 = 1,  H5 = 32,  W5 = 32;

  // Workspace carve-out (ping-pong + temp activation buffers sized for L1)
  char* ws = (char*)d_ws;
  const size_t BUF = (size_t)D1 * H1 * W1 * 16 * 2;  // 52.4 MB
  const size_t MSK = (size_t)D1 * H1 * W1 * 2;       // 3.3 MB
  bf16_t* bufA  = (bf16_t*)(ws);
  bf16_t* bufB  = (bf16_t*)(ws + BUF);
  bf16_t* bufT  = (bf16_t*)(ws + 2 * BUF);
  bf16_t* maskA = (bf16_t*)(ws + 3 * BUF);
  bf16_t* maskB = (bf16_t*)(ws + 3 * BUF + MSK);
  char*   wbase = ws + 3 * BUF + 2 * MSK;

  // 21 convs, each with its BN; d_in indices follow setup_inputs() insertion order
  struct CW { int widx, bnidx, bidx, Ci, Co, kd, kh, kw; };
  const CW cw[21] = {
    {  4,   5,  -1,   5,  16, 3,3,3},  // stem
    {  9,  11,  10,  16,  16, 3,3,3},  // c1_b0 conv1
    { 15,  17,  16,  16,  16, 3,3,3},  // c1_b0 conv2
    { 21,  23,  22,  16,  16, 3,3,3},  // c1_b1 conv1
    { 27,  29,  28,  16,  16, 3,3,3},  // c1_b1 conv2
    { 33,  34,  -1,  16,  32, 3,3,3},  // c2 down
    { 38,  40,  39,  32,  32, 3,3,3},
    { 44,  46,  45,  32,  32, 3,3,3},
    { 50,  52,  51,  32,  32, 3,3,3},
    { 56,  58,  57,  32,  32, 3,3,3},
    { 62,  63,  -1,  32,  64, 3,3,3},  // c3 down
    { 67,  69,  68,  64,  64, 3,3,3},
    { 73,  75,  74,  64,  64, 3,3,3},
    { 79,  81,  80,  64,  64, 3,3,3},
    { 85,  87,  86,  64,  64, 3,3,3},
    { 91,  92,  -1,  64, 128, 3,3,3},  // c4 down
    { 96,  98,  97, 128, 128, 3,3,3},
    {102, 104, 103, 128, 128, 3,3,3},
    {108, 110, 109, 128, 128, 3,3,3},
    {114, 116, 115, 128, 128, 3,3,3},
    {120, 121,  -1, 128, 128, 3,1,1},  // extra (3,1,1)
  };

  // Weight fragment slots + folded BN params in workspace
  bf16_t* wfp[21]; float* bns[21]; float* bnh[21];
  char* wc = wbase;
  for (int i = 0; i < 21; ++i) {
    const CW& c = cw[i];
    const int Cs = csz_host(c.Ci);
    const int m16 = (Cs == 16), kch = m16 ? 1 : Cs / 32, taps = c.kd * c.kh * c.kw;
    const int nst = m16 ? (taps + 1) / 2 : taps * kch;
    wfp[i] = (bf16_t*)wc;
    wc += (size_t)nst * (c.Co / 16) * 512 * 2;
  }
  float* bc = (float*)wc;
  for (int i = 0; i < 21; ++i) { bns[i] = bc; bc += cw[i].Co; bnh[i] = bc; bc += cw[i].Co; }

  // --- one-time-per-launch prep (deterministic) ---------------------------
  for (int i = 0; i < 21; ++i) {
    const CW& c = cw[i];
    const int Cs = csz_host(c.Ci);
    const int m16 = (Cs == 16), kch = m16 ? 1 : Cs / 32, taps = c.kd * c.kh * c.kw;
    const int nst = m16 ? (taps + 1) / 2 : taps * kch;
    const int coch = c.Co / 16, total = nst * coch * 512;
    wprep_k<<<dim3((total + 255) / 256), 256, 0, stream>>>(
        (const float*)d_in[c.widx], wfp[i], taps, c.Ci, c.Co, m16, kch, coch, nst);
    bnprep_k<<<dim3((c.Co + 63) / 64), 64, 0, stream>>>(
        (const float*)d_in[c.bnidx], (const float*)d_in[c.bnidx + 1],
        (const float*)d_in[c.bnidx + 2], (const float*)d_in[c.bnidx + 3],
        bns[i], bnh[i], c.Co);
  }

  // --- scatter ------------------------------------------------------------
  hipMemsetAsync(bufA, 0, BUF, stream);
  hipMemsetAsync(maskA, 0, MSK, stream);
  scatter_k<<<dim3((NV + 255) / 256), 256, 0, stream>>>(VF, COORS, bufA, maskA, NV, H1, W1);

  // --- conv launcher (dispatch to template instantiation) -----------------
  auto conv = [&](int i, const bf16_t* in, const bf16_t* mask, const bf16_t* skip,
                  bf16_t* out, int D, int H, int W, int OD, int OH, int OW,
                  int sd, int sh, int sw, int pd, int ph, int pw) {
    const CW& c = cw[i];
    ConvP p{};
    p.in = in; p.wf = wfp[i];
    p.bias  = (c.bidx >= 0) ? (const float*)d_in[c.bidx] : nullptr;
    p.scale = bns[i]; p.shift = bnh[i];
    p.mask = mask; p.skip = skip; p.out = out;
    p.D = D; p.H = H; p.W = W; p.OH = OH; p.OW = OW;
    p.Cout = c.Co; p.cochunks = c.Co / 16; p.lgwt = lg2_host(OW / 32);
    p.sd = sd; p.sh = sh; p.sw = sw; p.pd = pd; p.ph = ph; p.pw = pw;
    const int Cs = csz_host(c.Ci);
    const int KC = (Cs == 16) ? 1 : Cs / 32;
    dim3 g((unsigned)((OW / 32) * (c.Co / 16)), (unsigned)OH, (unsigned)OD);
    if (Cs == 16)        conv_wmma_t<1, 3, 3, 3, 1><<<g, 32, 0, stream>>>(p);
    else if (c.kw == 1)  conv_wmma_t<0, 3, 1, 1, 4><<<g, 32, 0, stream>>>(p);
    else if (KC == 1)    conv_wmma_t<0, 3, 3, 3, 1><<<g, 32, 0, stream>>>(p);
    else if (KC == 2)    conv_wmma_t<0, 3, 3, 3, 2><<<g, 32, 0, stream>>>(p);
    else                 conv_wmma_t<0, 3, 3, 3, 4><<<g, 32, 0, stream>>>(p);
  };
  auto pool = [&](const bf16_t* mi, bf16_t* mo, int D, int H, int W,
                  int OD, int OH, int OW, int kd, int kh, int kw,
                  int sd, int sh, int sw, int pd, int ph, int pw) {
    const int total = OD * OH * OW;
    maskpool_k<<<dim3((total + 255) / 256), 256, 0, stream>>>(
        mi, mo, D, H, W, OD, OH, OW, kd, kh, kw, sd, sh, sw, pd, ph, pw);
  };

  // --- network ------------------------------------------------------------
  // stem: relu(bn(conv(x)) * mask)
  conv(0, bufA, maskA, nullptr, bufB, D1,H1,W1, D1,H1,W1, 1,1,1, 1,1,1);
  // c1_b0 / c1_b1
  conv(1, bufB, maskA, nullptr, bufT, D1,H1,W1, D1,H1,W1, 1,1,1, 1,1,1);
  conv(2, bufT, maskA, bufB,    bufA, D1,H1,W1, D1,H1,W1, 1,1,1, 1,1,1);
  conv(3, bufA, maskA, nullptr, bufT, D1,H1,W1, D1,H1,W1, 1,1,1, 1,1,1);
  conv(4, bufT, maskA, bufA,    bufB, D1,H1,W1, D1,H1,W1, 1,1,1, 1,1,1);
  // c2 down + blocks
  pool(maskA, maskB, D1,H1,W1, D2,H2,W2, 3,3,3, 2,2,2, 1,1,1);
  conv(5, bufB, maskB, nullptr, bufA, D1,H1,W1, D2,H2,W2, 2,2,2, 1,1,1);
  conv(6, bufA, maskB, nullptr, bufT, D2,H2,W2, D2,H2,W2, 1,1,1, 1,1,1);
  conv(7, bufT, maskB, bufA,    bufB, D2,H2,W2, D2,H2,W2, 1,1,1, 1,1,1);
  conv(8, bufB, maskB, nullptr, bufT, D2,H2,W2, D2,H2,W2, 1,1,1, 1,1,1);
  conv(9, bufT, maskB, bufB,    bufA, D2,H2,W2, D2,H2,W2, 1,1,1, 1,1,1);
  // c3 down + blocks
  pool(maskB, maskA, D2,H2,W2, D3,H3,W3, 3,3,3, 2,2,2, 1,1,1);
  conv(10, bufA, maskA, nullptr, bufB, D2,H2,W2, D3,H3,W3, 2,2,2, 1,1,1);
  conv(11, bufB, maskA, nullptr, bufT, D3,H3,W3, D3,H3,W3, 1,1,1, 1,1,1);
  conv(12, bufT, maskA, bufB,    bufA, D3,H3,W3, D3,H3,W3, 1,1,1, 1,1,1);
  conv(13, bufA, maskA, nullptr, bufT, D3,H3,W3, D3,H3,W3, 1,1,1, 1,1,1);
  conv(14, bufT, maskA, bufA,    bufB, D3,H3,W3, D3,H3,W3, 1,1,1, 1,1,1);
  // c4 down (pad (0,1,1)) + blocks
  pool(maskA, maskB, D3,H3,W3, D4,H4,W4, 3,3,3, 2,2,2, 0,1,1);
  conv(15, bufB, maskB, nullptr, bufA, D3,H3,W3, D4,H4,W4, 2,2,2, 0,1,1);
  conv(16, bufA, maskB, nullptr, bufT, D4,H4,W4, D4,H4,W4, 1,1,1, 1,1,1);
  conv(17, bufT, maskB, bufA,    bufB, D4,H4,W4, D4,H4,W4, 1,1,1, 1,1,1);
  conv(18, bufB, maskB, nullptr, bufT, D4,H4,W4, D4,H4,W4, 1,1,1, 1,1,1);
  conv(19, bufT, maskB, bufB,    bufA, D4,H4,W4, D4,H4,W4, 1,1,1, 1,1,1);
  // extra: kernel (3,1,1), stride (2,1,1), no pad
  pool(maskB, maskA, D4,H4,W4, D5,H5,W5, 3,1,1, 2,1,1, 0,0,0);
  conv(20, bufA, maskA, nullptr, bufB, D4,H4,W4, D5,H5,W5, 2,1,1, 0,0,0);

  // (1,32,32,128) bf16 -> (128,32,32) fp32
  finalize_k<<<dim3((128 * H5 * W5 + 255) / 256), 256, 0, stream>>>(
      bufB, (float*)d_out, 128, H5 * W5);
}
